// ExtGNNLayer_386547057414
// MI455X (gfx1250) — compile-verified, matrix operand order checked by reading, not run on hardware
//
#include <hip/hip_runtime.h>
#include <hip/hip_bf16.h>

// ---------------------------------------------------------------------------
// Problem constants (from the reference)
// ---------------------------------------------------------------------------
#define NN 50000      // nodes
#define EE 500000     // edges
#define RR 500        // relations
#define TT 366        // times
#define DD 128        // dim
#define IN3 384       // 3*D

typedef __attribute__((ext_vector_type(16))) __bf16 v16bf;
typedef __attribute__((ext_vector_type(8)))  float  v8f;
typedef __attribute__((ext_vector_type(4)))  unsigned int u32x4;
typedef __attribute__((ext_vector_type(8)))  int    i32x8;
typedef __attribute__((ext_vector_type(4)))  int    i32x4;

union BfVec {
  unsigned       d[8];
  uint4          q[2];
  v16bf          v;
};

// Pack two fp32 -> packed bf16 pair (round-half-up + v_perm_b32): 3 VALU per pair.
__device__ __forceinline__ unsigned pack_bf16(float f0, float f1) {
  unsigned u0 = __builtin_bit_cast(unsigned, f0) + 0x8000u;
  unsigned u1 = __builtin_bit_cast(unsigned, f1) + 0x8000u;
  return __builtin_amdgcn_perm(u1, u0, 0x07060302u);  // {u1.hi16, u0.hi16}
}

__device__ __forceinline__ unsigned short f2bf_bits(float f) {
  unsigned u = __builtin_bit_cast(unsigned, f);
  u += 0x7FFFu + ((u >> 16) & 1u);   // round-to-nearest-even
  return (unsigned short)(u >> 16);
}

// ---------------------------------------------------------------------------
// Tensor Data Mover: 2D tile load (bf16 elements) into LDS.
//   D# group0: count=1 | lds_addr | global_addr | type=2
//   D# group1: data_size=2B, tensor_dim0/1, tile_dim0/1 (tile_dim2=0 -> 2D),
//              tensor_dim0_stride. Groups 2/3 unused for 2D -> zeros.
// This toolchain exposes the 6-arg builtin:
//   (u32x4 g0, i32x8 g1, i32x4, i32x4, i32x8, i32 cpol)
// Issued by one wave (TDM ignores EXEC, tracked with TENSORcnt).
// NOTE: the builtin does not take an LDS pointer, so after the wait we must
// tell the compiler the LDS array was written (inline-asm memory clobber),
// otherwise it proves the LDS reads undef and deletes the ds_loads + WMMAs.
// ---------------------------------------------------------------------------
#if __has_builtin(__builtin_amdgcn_tensor_load_to_lds) && __has_builtin(__builtin_amdgcn_s_wait_tensorcnt)
#define HAVE_TDM 1
#else
#define HAVE_TDM 0
#endif

#if HAVE_TDM
__device__ __forceinline__ void tdm_load_2d_bf16(const void* gsrc, unsigned lds_byte_off,
                                                 unsigned tile_d0, unsigned tile_d1,
                                                 unsigned tensor_d0, unsigned tensor_d1,
                                                 unsigned stride_d0) {
  unsigned long long ga = (unsigned long long)gsrc;
  u32x4 g0;
  g0[0] = 1u;                                            // count=1, user descriptor
  g0[1] = lds_byte_off;                                  // lds_addr (bytes)
  g0[2] = (unsigned)(ga & 0xFFFFFFFFull);                // global_addr[31:0]
  g0[3] = (unsigned)((ga >> 32) & 0x01FFFFFFull)         // global_addr[56:32]
        | (2u << 30);                                    // type=2 ("image")
  union { unsigned w[8]; i32x8 v; } g1;
  g1.w[0] = (1u << 16);                                  // data_size=1 (2 bytes)
  g1.w[1] = (tensor_d0 & 0xFFFFu) << 16;                 // tensor_dim0[15:0]
  g1.w[2] = (tensor_d0 >> 16) | ((tensor_d1 & 0xFFFFu) << 16);
  g1.w[3] = (tensor_d1 >> 16) | (tile_d0 << 16);         // tile_dim0
  g1.w[4] = tile_d1;                                     // tile_dim1 (tile_dim2=0)
  g1.w[5] = stride_d0;                                   // tensor_dim0_stride[31:0]
  g1.w[6] = 0;
  g1.w[7] = 0;
  i32x4 z4 = (i32x4){0, 0, 0, 0};
  i32x8 z8 = (i32x8){0, 0, 0, 0, 0, 0, 0, 0};
  __builtin_amdgcn_tensor_load_to_lds(g0, g1.v, z4, z4, z8, 0);
}
#endif

// ---------------------------------------------------------------------------
// Workspace layout (bytes)
// ---------------------------------------------------------------------------
static const size_t OFF_SUMS = 0;                       // NN*DD fp32   = 25,600,000 B
static const size_t OFF_CNT  = 25600000;                // NN   fp32    =    200,000 B
static const size_t OFF_WIT  = 25800192;                // 128x384 bf16 =     98,304 B
static const size_t OFF_WOT  = OFF_WIT + 98304;
static const size_t OFF_WST  = OFF_WOT + 98304;         // 128x128 bf16 =     32,768 B
// total ~26.03 MB

// ---------------------------------------------------------------------------
// Zero workspace (sums + cnt are contiguous)
// ---------------------------------------------------------------------------
__global__ void zero_kernel(float* __restrict__ p, int n) {
  int stride = gridDim.x * blockDim.x;
  for (int i = blockIdx.x * blockDim.x + threadIdx.x; i < n; i += stride)
    p[i] = 0.0f;
}

// ---------------------------------------------------------------------------
// Transpose + convert weights to bf16: Wt[n][k] = bf16(W[k][n])
// ---------------------------------------------------------------------------
__global__ void prep_kernel(const float* __restrict__ WI, const float* __restrict__ WO,
                            const float* __restrict__ WS,
                            unsigned short* __restrict__ WIt,
                            unsigned short* __restrict__ WOt,
                            unsigned short* __restrict__ WSt) {
  int i = blockIdx.x * blockDim.x + threadIdx.x;      // grid covers 49152 exactly
  if (i < DD * IN3) {
    int n = i / IN3, k = i % IN3;
    WIt[i] = f2bf_bits(WI[(size_t)k * DD + n]);
    WOt[i] = f2bf_bits(WO[(size_t)k * DD + n]);
  }
  if (i < DD * DD) {
    int n = i / DD, k = i % DD;
    WSt[i] = f2bf_bits(WS[(size_t)k * DD + n]);
  }
}

// ---------------------------------------------------------------------------
// Per-destination edge counts (for mean aggregation)
// ---------------------------------------------------------------------------
__global__ void count_kernel(const int* __restrict__ dst, float* __restrict__ cnt) {
  int e = blockIdx.x * blockDim.x + threadIdx.x;
  if (e < EE) atomicAdd(cnt + dst[e], 1.0f);
}

// ---------------------------------------------------------------------------
// Edge message kernel: one wave per 16-edge tile.
//   comp = [rel_emb | ent_emb(src) | time_emb]  (K = 384)
//   msg_i = comp @ WI, msg_o = comp @ WO  via v_wmma_f32_16x16x32_bf16
//   select by inv, add bias, atomic-add into sums[dst].
// Transposed bf16 weights staged through LDS in 4 K-slices of 96 via the TDM.
// ---------------------------------------------------------------------------
__global__ __launch_bounds__(256) void edge_msg_kernel(
    const int* __restrict__ src, const int* __restrict__ dst,
    const int* __restrict__ b_rel, const int* __restrict__ time_ids,
    const int* __restrict__ inv,
    const float* __restrict__ ent_emb, const float* __restrict__ rel_emb,
    const float* __restrict__ time_emb,
    const unsigned short* __restrict__ WIt, const unsigned short* __restrict__ WOt,
    const float* __restrict__ WI_b, const float* __restrict__ WO_b,
    float* __restrict__ sums) {
  __shared__ unsigned short sW[2 * 128 * 96];  // 48 KB, only __shared__ var -> LDS offset 0

  const int lane = threadIdx.x & 31;
  const int wave = threadIdx.x >> 5;
  const int tile = blockIdx.x * 8 + wave;
  const bool valid = (tile < (EE / 16));
  const int m    = lane & 15;                         // row within tile / col within n-tile
  const int half = lane >> 4;                         // K-group selector

  int relI = 0, srcI = 0, timI = 0;
  if (valid) {
    int e = tile * 16 + m;
    relI = b_rel[e]; srcI = src[e]; timI = time_ids[e];
  }
  const float* rowR = rel_emb  + (size_t)relI * DD;
  const float* rowS = ent_emb  + (size_t)srcI * DD;
  const float* rowT = time_emb + (size_t)timI * DD;

  v8f accI[8], accO[8];
  #pragma unroll
  for (int nt = 0; nt < 8; ++nt) {
    accI[nt] = (v8f){0,0,0,0,0,0,0,0};
    accO[nt] = (v8f){0,0,0,0,0,0,0,0};
  }

  #pragma unroll
  for (int p = 0; p < 4; ++p) {
    __syncthreads();                                   // previous slice fully consumed
#if HAVE_TDM
    if (wave == 0) {                                   // wave-uniform; TDM ignores EXEC
      tdm_load_2d_bf16(WIt + p * 96, /*lds*/ 0,     96, 128, IN3, 128, IN3);
      tdm_load_2d_bf16(WOt + p * 96, /*lds*/ 24576, 96, 128, IN3, 128, IN3);
      __builtin_amdgcn_s_wait_tensorcnt(0);
      // TDM wrote sW behind the compiler's back -> declare it modified.
      asm volatile("" : : "v"(&sW[0]) : "memory");
    }
#else
    {                                                  // manual staging fallback
      int r  = threadIdx.x >> 1;                       // 0..127
      int hv = threadIdx.x & 1;
      #pragma unroll
      for (int mat = 0; mat < 2; ++mat) {
        const uint4* s4 = (const uint4*)((mat ? WOt : WIt) + (size_t)r * IN3 + p * 96 + hv * 48);
        uint4* d4 = (uint4*)(sW + mat * (128 * 96) + r * 96 + hv * 48);
        #pragma unroll
        for (int i = 0; i < 6; ++i) d4[i] = s4[i];
      }
    }
#endif
    __syncthreads();

    if (valid) {
      #pragma unroll
      for (int sl = 0; sl < 3; ++sl) {
        const int kb   = 32 * (3 * p + sl);            // global K base (const-folded)
        const int tsel = kb >> 7;                      // which embedding table
        const int off  = kb & 127;
        const float* tab = (tsel == 0) ? rowR : (tsel == 1) ? rowS : rowT;
        // A fragment: lane holds row m; half selects K-groups per 16-bit A layout.
        const float* q0 = tab + off + 8 * half;        // K = kb + 8*half + [0..8)
        const float* q1 = tab + off + 16 + 8 * half;   // K = kb + 16 + 8*half + [0..8)
        float4 c0a = ((const float4*)q0)[0], c0b = ((const float4*)q0)[1];
        float4 c1a = ((const float4*)q1)[0], c1b = ((const float4*)q1)[1];
        BfVec A;
        A.d[0] = pack_bf16(c0a.x, c0a.y);  A.d[1] = pack_bf16(c0a.z, c0a.w);
        A.d[2] = pack_bf16(c0b.x, c0b.y);  A.d[3] = pack_bf16(c0b.z, c0b.w);
        A.d[4] = pack_bf16(c1a.x, c1a.y);  A.d[5] = pack_bf16(c1a.z, c1a.w);
        A.d[6] = pack_bf16(c1b.x, c1b.y);  A.d[7] = pack_bf16(c1b.z, c1b.w);

        const int kl = 32 * sl;                        // K base within LDS slice
        #pragma unroll
        for (int nt = 0; nt < 8; ++nt) {
          const unsigned short* bI = sW + (nt * 16 + m) * 96 + kl + 8 * half;
          BfVec BI;
          BI.q[0] = *(const uint4*)bI;
          BI.q[1] = *(const uint4*)(bI + 16);
          accI[nt] = __builtin_amdgcn_wmma_f32_16x16x32_bf16(
              false, A.v, false, BI.v, (short)0, accI[nt], false, false);
          const unsigned short* bO = bI + 128 * 96;
          BfVec BO;
          BO.q[0] = *(const uint4*)bO;
          BO.q[1] = *(const uint4*)(bO + 16);
          accO[nt] = __builtin_amdgcn_wmma_f32_16x16x32_bf16(
              false, A.v, false, BO.v, (short)0, accO[nt], false, false);
        }
      }
    }
  }

  if (valid) {
    // C/D layout: acc VGPR j holds row (j + 8*half), column m of the n-tile.
    int invr[8], dstr[8];
    #pragma unroll
    for (int j = 0; j < 8; ++j) {
      int er = tile * 16 + j + 8 * half;
      invr[j] = inv[er];
      dstr[j] = dst[er];
    }
    #pragma unroll
    for (int nt = 0; nt < 8; ++nt) {
      int col = nt * 16 + m;
      float bIv = WI_b[col], bOv = WO_b[col];
      #pragma unroll
      for (int j = 0; j < 8; ++j) {
        float v = invr[j] ? (accO[nt][j] + bOv) : (accI[nt][j] + bIv);
        atomicAdd(sums + (size_t)dstr[j] * DD + col, v);
      }
    }
  }
}

// ---------------------------------------------------------------------------
// Node update: ent_new = ent_emb @ WS + WS_b + sums / max(cnt,1)
// ---------------------------------------------------------------------------
__global__ __launch_bounds__(256) void node_update_kernel(
    const float* __restrict__ ent_emb, const unsigned short* __restrict__ WSt,
    const float* __restrict__ WS_b, const float* __restrict__ sums,
    const float* __restrict__ cnt, float* __restrict__ out) {
  __shared__ unsigned short sW[128 * 128];  // 32 KB, only __shared__ var -> LDS offset 0
#if HAVE_TDM
  if ((threadIdx.x >> 5) == 0) {
    tdm_load_2d_bf16(WSt, /*lds*/ 0, 128, 128, 128, 128, 128);
    __builtin_amdgcn_s_wait_tensorcnt(0);
    // TDM wrote sW behind the compiler's back -> declare it modified.
    asm volatile("" : : "v"(&sW[0]) : "memory");
  }
#else
  {
    int r = threadIdx.x >> 1, hv = threadIdx.x & 1;
    const uint4* s4 = (const uint4*)(WSt + (size_t)r * DD + hv * 64);
    uint4* d4 = (uint4*)(sW + r * DD + hv * 64);
    #pragma unroll
    for (int i = 0; i < 8; ++i) d4[i] = s4[i];
  }
#endif
  __syncthreads();

  const int lane = threadIdx.x & 31;
  const int wave = threadIdx.x >> 5;
  const int tile = blockIdx.x * 8 + wave;
  if (tile >= NN / 16) return;                         // no barriers past this point
  const int m = lane & 15, half = lane >> 4;

  const float* arow = ent_emb + (size_t)(tile * 16 + m) * DD;
  v8f acc[8];
  #pragma unroll
  for (int nt = 0; nt < 8; ++nt) acc[nt] = (v8f){0,0,0,0,0,0,0,0};

  #pragma unroll
  for (int ks = 0; ks < 4; ++ks) {
    const int kb = 32 * ks;
    const float* q0 = arow + kb + 8 * half;
    const float* q1 = arow + kb + 16 + 8 * half;
    float4 c0a = ((const float4*)q0)[0], c0b = ((const float4*)q0)[1];
    float4 c1a = ((const float4*)q1)[0], c1b = ((const float4*)q1)[1];
    BfVec A;
    A.d[0] = pack_bf16(c0a.x, c0a.y);  A.d[1] = pack_bf16(c0a.z, c0a.w);
    A.d[2] = pack_bf16(c0b.x, c0b.y);  A.d[3] = pack_bf16(c0b.z, c0b.w);
    A.d[4] = pack_bf16(c1a.x, c1a.y);  A.d[5] = pack_bf16(c1a.z, c1a.w);
    A.d[6] = pack_bf16(c1b.x, c1b.y);  A.d[7] = pack_bf16(c1b.z, c1b.w);
    #pragma unroll
    for (int nt = 0; nt < 8; ++nt) {
      const unsigned short* b = sW + (nt * 16 + m) * DD + kb + 8 * half;
      BfVec B;
      B.q[0] = *(const uint4*)b;
      B.q[1] = *(const uint4*)(b + 16);
      acc[nt] = __builtin_amdgcn_wmma_f32_16x16x32_bf16(
          false, A.v, false, B.v, (short)0, acc[nt], false, false);
    }
  }

  int   noder[8];
  float rden[8];
  #pragma unroll
  for (int j = 0; j < 8; ++j) {
    int node = tile * 16 + j + 8 * half;
    noder[j] = node;
    rden[j]  = 1.0f / fmaxf(cnt[node], 1.0f);
  }
  #pragma unroll
  for (int nt = 0; nt < 8; ++nt) {
    int col = nt * 16 + m;
    float bs = WS_b[col];
    #pragma unroll
    for (int j = 0; j < 8; ++j) {
      size_t idx = (size_t)noder[j] * DD + col;
      out[idx] = acc[nt][j] + bs + sums[idx] * rden[j];
    }
  }
}

// ---------------------------------------------------------------------------
// Small tables: rel_new, time_new (plain fp32, negligible work)
// ---------------------------------------------------------------------------
__global__ void reltime_kernel(const float* __restrict__ rel_emb,
                               const float* __restrict__ time_emb,
                               const float* __restrict__ WR_w, const float* __restrict__ WR_b,
                               const float* __restrict__ WT_w, const float* __restrict__ WT_b,
                               float* __restrict__ out) {
  int idx = blockIdx.x * blockDim.x + threadIdx.x;
  const int RN = RR * DD, TN = TT * DD;
  if (idx < RN) {
    int r = idx >> 7, n = idx & 127;
    float a = WR_b[n];
    for (int k = 0; k < DD; ++k) a += rel_emb[(size_t)r * DD + k] * WR_w[(size_t)k * DD + n];
    out[(size_t)NN * DD + idx] = a;
  } else if (idx < RN + TN) {
    int j = idx - RN, t = j >> 7, n = j & 127;
    float a = WT_b[n];
    for (int k = 0; k < DD; ++k) a += time_emb[(size_t)t * DD + k] * WT_w[(size_t)k * DD + n];
    out[(size_t)NN * DD + RN + j] = a;
  }
}

// ---------------------------------------------------------------------------
// Host launcher
// ---------------------------------------------------------------------------
extern "C" void kernel_launch(void* const* d_in, const int* in_sizes, int n_in,
                              void* d_out, int out_size, void* d_ws, size_t ws_size,
                              hipStream_t stream) {
  const int*   src      = (const int*)  d_in[0];
  const int*   dst      = (const int*)  d_in[1];
  const int*   b_rel    = (const int*)  d_in[2];
  const int*   time_ids = (const int*)  d_in[3];
  const int*   inv      = (const int*)  d_in[4];
  const float* ent_emb  = (const float*)d_in[5];
  const float* rel_emb  = (const float*)d_in[6];
  const float* time_emb = (const float*)d_in[7];
  const float* WI_w     = (const float*)d_in[8];
  const float* WI_b     = (const float*)d_in[9];
  const float* WO_w     = (const float*)d_in[10];
  const float* WO_b     = (const float*)d_in[11];
  const float* WS_w     = (const float*)d_in[12];
  const float* WS_b     = (const float*)d_in[13];
  const float* WR_w     = (const float*)d_in[14];
  const float* WR_b     = (const float*)d_in[15];
  const float* WT_w     = (const float*)d_in[16];
  const float* WT_b     = (const float*)d_in[17];

  char* ws = (char*)d_ws;
  float*          sums = (float*)(ws + OFF_SUMS);
  float*          cnt  = (float*)(ws + OFF_CNT);
  unsigned short* WIt  = (unsigned short*)(ws + OFF_WIT);
  unsigned short* WOt  = (unsigned short*)(ws + OFF_WOT);
  unsigned short* WSt  = (unsigned short*)(ws + OFF_WST);
  float* out = (float*)d_out;

  // 1) zero accumulators (sums + cnt contiguous)
  zero_kernel<<<4096, 256, 0, stream>>>(sums, NN * DD + NN);
  // 2) transpose+convert weights to bf16 (49152 threads covers all three)
  prep_kernel<<<(DD * IN3 + 255) / 256, 256, 0, stream>>>(WI_w, WO_w, WS_w, WIt, WOt, WSt);
  // 3) per-destination counts
  count_kernel<<<(EE + 255) / 256, 256, 0, stream>>>(dst, cnt);
  // 4) edge messages (WMMA, TDM-staged weights) + scatter
  {
    int tiles = EE / 16;                 // 31250
    int blocks = (tiles + 7) / 8;        // 3907
    edge_msg_kernel<<<blocks, 256, 0, stream>>>(src, dst, b_rel, time_ids, inv,
                                                ent_emb, rel_emb, time_emb,
                                                WIt, WOt, WI_b, WO_b, sums);
  }
  // 5) node update (WMMA) fused with mean aggregation
  {
    int tiles = NN / 16;                 // 3125
    int blocks = (tiles + 7) / 8;        // 391
    node_update_kernel<<<blocks, 256, 0, stream>>>(ent_emb, WSt, WS_b, sums, cnt, out);
  }
  // 6) rel/time table updates
  reltime_kernel<<<((RR + TT) * DD + 255) / 256, 256, 0, stream>>>(
      rel_emb, time_emb, WR_w, WR_b, WT_w, WT_b, out);
}
